// CGMC_64072322122515
// MI455X (gfx1250) — compile-verified, matrix operand order checked by reading, not run on hardware
//
#include <hip/hip_runtime.h>
#include <hip/hip_bf16.h>
#include <math.h>

typedef __attribute__((ext_vector_type(16))) _Float16 v16h;
typedef __attribute__((ext_vector_type(8)))  float    v8f;
typedef unsigned int v4u __attribute__((ext_vector_type(4)));
typedef int          v8i __attribute__((ext_vector_type(8)));
typedef int          v4i __attribute__((ext_vector_type(4)));

#define WPB  4    // waves per block in WMMA kernels (4 x wave32 = 128 threads)
#define EROW 65   // padded LDS row stride (DWORDs) for 64-wide efeats tile: bank-conflict-free
#define ZROW 129  // padded LDS row stride (DWORDs) for 128-wide head tile: bank-conflict-free

#ifndef __has_builtin
#define __has_builtin(x) 0
#endif
#if __has_builtin(__builtin_amdgcn_tensor_load_to_lds)
#define HAVE_TDM 1
#else
#define HAVE_TDM 0
#endif

__device__ __forceinline__ float sigmoidf_(float x){ return 1.f/(1.f+expf(-x)); }
__device__ __forceinline__ float eluf_(float x){ return x>0.f ? x : expm1f(x); }
__device__ __forceinline__ float lreluf_(float x){ return x>0.f ? x : 0.01f*x; }
// order-preserving float <-> uint map for atomic max on signed floats
__device__ __forceinline__ unsigned flipf_(float f){
    unsigned u=__float_as_uint(f); return (u&0x80000000u) ? ~u : (u|0x80000000u);
}
__device__ __forceinline__ float unflipf_(unsigned u){
    return __uint_as_float((u&0x80000000u) ? (u&0x7FFFFFFFu) : ~u);
}
// K index pattern for 16-bit 16x32 A (and mirrored 32x16 B) fragments, ISA 7.12.2
__device__ __forceinline__ int kpat_(int v,int half){
    return ((v>>2)<<4) + (half<<3) + ((v&3)<<1);
}

__global__ void k_zero(float* __restrict__ p, long n){
    long i = (long)blockIdx.x*blockDim.x + threadIdx.x;
    if(i<n) p[i]=0.f;
}

// h = x@Wn ; s_src/s_dst = (h*a).sum over D per head
__global__ void k_node_proj(const float* __restrict__ x, const float* __restrict__ Wn,
                            const float* __restrict__ a_src, const float* __restrict__ a_dst,
                            float* __restrict__ h, float* __restrict__ ssrc,
                            float* __restrict__ sdst, int N)
{
    int n = blockIdx.x*blockDim.x + threadIdx.x;
    if(n>=N) return;
    float xv[4];
    #pragma unroll
    for(int k=0;k<4;k++) xv[k]=x[(long)n*4+k];
    float hv[32];
    #pragma unroll
    for(int j=0;j<32;j++){
        float acc=0.f;
        #pragma unroll
        for(int k=0;k<4;k++) acc += xv[k]*Wn[k*32+j];
        hv[j]=acc; h[(long)n*32+j]=acc;
    }
    #pragma unroll
    for(int hh=0;hh<4;hh++){
        float ss=0.f, sd=0.f;
        #pragma unroll
        for(int d=0;d<8;d++){ ss+=hv[hh*8+d]*a_src[hh*8+d]; sd+=hv[hh*8+d]*a_dst[hh*8+d]; }
        ssrc[(long)n*4+hh]=ss; sdst[(long)n*4+hh]=sd;
    }
}

// per 16-edge tile: TDM-stage efeats tile into padded LDS, e_proj = efeats@We + be via WMMA,
// then fused e_sig=sigmoid(e_proj), att=leaky(s_src[src]+s_dst[dst]+e_proj@Wae), segment-max
__global__ void k_edge_wmma(const float* __restrict__ efeats, const float* __restrict__ We,
                            const float* __restrict__ be, const float* __restrict__ Wae,
                            const int* __restrict__ src, const int* __restrict__ dst,
                            const float* __restrict__ ssrc, const float* __restrict__ sdst,
                            float* __restrict__ esig, float* __restrict__ att,
                            unsigned* __restrict__ mkey, int ntiles, int E)
{
    __shared__ float T[WPB*16*EROW];   // TDM-padded efeats tiles
    __shared__ float S[WPB*16*8];      // e_proj staging
    int wave = threadIdx.x>>5, lane = threadIdx.x&31;
    int tile = blockIdx.x*WPB + wave;
    if(tile>=ntiles) return;                 // wave-uniform: EXEC stays all-ones
    float* Tw = T + wave*(16*EROW);
    const float* gsrc = efeats + (long)tile*16*64;
    if(tile+WPB < ntiles) __builtin_prefetch(gsrc + (long)WPB*16*64, 0, 0);

#if HAVE_TDM
    {
        // D# per ISA 8.3/8.4: 2D tensor, 4B elements, tile 64x16,
        // pad 1 DWORD every 64 DWORDs -> LDS row stride 65 (bank-conflict-free reads)
        unsigned ldsOff = (unsigned)(size_t)Tw;
        unsigned long long ga = (unsigned long long)(const void*)gsrc;
        v4u g0 = { 1u,                                   // count=1, user descriptor
                   ldsOff,                               // lds_addr
                   (unsigned)(ga & 0xFFFFFFFFu),         // global_addr[31:0]
                   (unsigned)(((ga>>32)&0x1FFFFFFu) | 0x80000000u) }; // addr[56:32] | type=2
        v8i g1 = { (int)((2u<<16)|(1u<<20)|(5u<<22)),    // data_size=4B, pad_en, interval=64DW
                   (int)(64u<<16),                       // tensor_dim0 = 64
                   (int)(((unsigned)E & 0xFFFFu)<<16),   // tensor_dim1 lo16 = E
                   (int)(((unsigned)E >> 16) | (64u<<16)), // tensor_dim1 hi16 | tile_dim0=64
                   16,                                   // tile_dim1 = 16 rows
                   64,                                   // tensor_dim0_stride = 64
                   0, 0 };
        v4i gz = { 0,0,0,0 };
#if __clang_major__ >= 23
        v8i gz8 = { 0,0,0,0,0,0,0,0 };
        __builtin_amdgcn_tensor_load_to_lds(g0, g1, gz, gz, gz8, 0);
#else
        __builtin_amdgcn_tensor_load_to_lds(g0, g1, gz, gz, 0);
#endif
    }
#else
    // fallback: coalesced cooperative stage into the same padded layout
    for(int i=lane;i<1024;i+=32){
        int row=i>>6, col=i&63;
        Tw[row*EROW+col]=gsrc[i];
    }
#endif

    // B fragments from We (64x8 padded to 64x16) — overlaps with the tile transfer
    int m = lane&15, half = lane>>4, n = m;
    bool vn = (n<8);
    v16h b0,b1;
    #pragma unroll
    for(int v=0;v<8;v++){
        int k0 = kpat_(v,half);
        b0[2*v]  = vn ? (_Float16)We[(k0  )*8+n] : (_Float16)0.f;
        b0[2*v+1]= vn ? (_Float16)We[(k0+1)*8+n] : (_Float16)0.f;
        b1[2*v]  = vn ? (_Float16)We[(32+k0)*8+n] : (_Float16)0.f;
        b1[2*v+1]= vn ? (_Float16)We[(33+k0)*8+n] : (_Float16)0.f;
    }

#if HAVE_TDM
#if __has_builtin(__builtin_amdgcn_s_wait_tensorcnt)
    __builtin_amdgcn_s_wait_tensorcnt(0);
#else
    asm volatile("s_wait_tensorcnt 0x0" ::: "memory");
#endif
#endif
    asm volatile("s_wait_dscnt 0" ::: "memory");  // order cross-lane LDS stage vs reads

    const float* arow = Tw + m*EROW;
    v16h a0,a1;
    #pragma unroll
    for(int v=0;v<8;v++){
        int k0 = kpat_(v,half);
        a0[2*v]=(_Float16)arow[k0];      a0[2*v+1]=(_Float16)arow[k0+1];
        a1[2*v]=(_Float16)arow[32+k0];   a1[2*v+1]=(_Float16)arow[32+k0+1];
    }
    v8f c = {};
    c = __builtin_amdgcn_wmma_f32_16x16x32_f16(false,a0,false,b0,(short)0,c,false,false);
    c = __builtin_amdgcn_wmma_f32_16x16x32_f16(false,a1,false,b1,(short)0,c,false,false);

    float* Sw = S + wave*128;                // wave-private [16 edges][8 rel]
    if(vn){
        float bias = be[n];
        #pragma unroll
        for(int r=0;r<8;r++) Sw[(half*8+r)*8+n] = c[r]+bias;
    }
    asm volatile("s_wait_dscnt 0" ::: "memory");
    if(lane<16){
        long e = (long)tile*16 + lane;
        float ep[8];
        #pragma unroll
        for(int r=0;r<8;r++) ep[r]=Sw[lane*8+r];
        #pragma unroll
        for(int r=0;r<8;r++) esig[e*8+r] = sigmoidf_(ep[r]);
        int s = src[e], d = dst[e];
        #pragma unroll
        for(int hh=0;hh<4;hh++){
            float ea=0.f;
            #pragma unroll
            for(int r=0;r<8;r++) ea += ep[r]*Wae[r*4+hh];
            float av = lreluf_(ssrc[(long)s*4+hh] + sdst[(long)d*4+hh] + ea);
            att[e*4+hh]=av;
            atomicMax(&mkey[(long)d*4+hh], flipf_(av));
        }
    }
}

// per (edge,head): ex = exp(att-m[dst]); segment-sum
__global__ void k_expsum(const float* __restrict__ att, const int* __restrict__ dst,
                         const unsigned* __restrict__ mkey, float* __restrict__ ssum, long n4)
{
    long t = (long)blockIdx.x*blockDim.x + threadIdx.x;
    if(t>=n4) return;
    long e=t>>2; int hh=t&3; int d=dst[e];
    float ex = expf(att[t] - unflipf_(mkey[(long)d*4+hh]));
    atomicAdd(&ssum[(long)d*4+hh], ex);
}

// per (edge,head): alpha*h[src] scattered into x1[dst]
__global__ void k_scatter1(const float* __restrict__ att, const int* __restrict__ src,
                           const int* __restrict__ dst, const unsigned* __restrict__ mkey,
                           const float* __restrict__ ssum, const float* __restrict__ mask,
                           const float* __restrict__ h, float* __restrict__ x1, long n4)
{
    long t = (long)blockIdx.x*blockDim.x + threadIdx.x;
    if(t>=n4) return;
    long e=t>>2; int hh=t&3; int d=dst[e], s=src[e];
    float ex = expf(att[t] - unflipf_(mkey[(long)d*4+hh]));
    float alpha = ex/(ssum[(long)d*4+hh]+1e-9f)*mask[e];
    #pragma unroll
    for(int dd=0;dd<8;dd++)
        atomicAdd(&x1[(long)d*32+hh*8+dd], alpha*h[(long)s*32+hh*8+dd]);
}

__global__ void k_elu_inplace(float* __restrict__ p, long n){
    long i = (long)blockIdx.x*blockDim.x + threadIdx.x;
    if(i<n) p[i]=eluf_(p[i]);
}

// per 16-node tile: hr[n,t,:] = x1@Wrel[t] (6 WMMAs), xself = x1@Wself + bself (2 WMMAs)
__global__ void k_node_wmma(const float* __restrict__ x1, const float* __restrict__ Wrel,
                            const float* __restrict__ Wself, const float* __restrict__ bself,
                            float* __restrict__ hr, float* __restrict__ xself, int ntiles)
{
    int wave=threadIdx.x>>5, lane=threadIdx.x&31;
    int tile=blockIdx.x*WPB+wave;
    if(tile>=ntiles) return;
    int m=lane&15, half=lane>>4, n=m;
    const float* xrow = x1 + ((long)tile*16+m)*32;
    v16h a;
    #pragma unroll
    for(int v=0;v<8;v++){ int k0=kpat_(v,half);
        a[2*v]=(_Float16)xrow[k0]; a[2*v+1]=(_Float16)xrow[k0+1]; }
    #pragma unroll
    for(int t=0;t<3;t++){
        #pragma unroll
        for(int nb=0;nb<2;nb++){
            v16h b; int col=nb*16+n;
            #pragma unroll
            for(int v=0;v<8;v++){ int k0=kpat_(v,half);
                b[2*v]  =(_Float16)Wrel[t*1024+(k0  )*32+col];
                b[2*v+1]=(_Float16)Wrel[t*1024+(k0+1)*32+col]; }
            v8f c={};
            c=__builtin_amdgcn_wmma_f32_16x16x32_f16(false,a,false,b,(short)0,c,false,false);
            #pragma unroll
            for(int r=0;r<8;r++){
                long node=(long)tile*16+half*8+r;
                hr[node*96 + t*32 + col]=c[r];
            }
        }
    }
    #pragma unroll
    for(int nb=0;nb<2;nb++){
        v16h b; int col=nb*16+n;
        #pragma unroll
        for(int v=0;v<8;v++){ int k0=kpat_(v,half);
            b[2*v]  =(_Float16)Wself[(k0  )*32+col];
            b[2*v+1]=(_Float16)Wself[(k0+1)*32+col]; }
        v8f c={};
        c=__builtin_amdgcn_wmma_f32_16x16x32_f16(false,a,false,b,(short)0,c,false,false);
        float bia=bself[col];
        #pragma unroll
        for(int r=0;r<8;r++){
            long node=(long)tile*16+half*8+r;
            xself[node*32+col]=c[r]+bia;
        }
    }
}

// per (edge, 4-output eighth): gate recomputed from e_sig, scatter hr[src,etype]*gate into agg2[dst]
__global__ void k_scatter2(const float* __restrict__ esig, const float* __restrict__ Wef,
                           const float* __restrict__ hr, const int* __restrict__ src,
                           const int* __restrict__ dst, const int* __restrict__ etype,
                           const float* __restrict__ mask,
                           float* __restrict__ agg2, float* __restrict__ deg, long n8)
{
    long t = (long)blockIdx.x*blockDim.x + threadIdx.x;
    if(t>=n8) return;
    long e=t>>3; int oq=t&7;
    int d=dst[e], s=src[e], et=etype[e];
    float mk=mask[e];
    if(oq==0) atomicAdd(&deg[d], mk);
    float es[8];
    #pragma unroll
    for(int r=0;r<8;r++) es[r]=esig[e*8+r];
    #pragma unroll
    for(int j=0;j<4;j++){
        int o=oq*4+j;
        float g=0.f;
        #pragma unroll
        for(int r=0;r<8;r++) g+=es[r]*Wef[r*32+o];
        g=sigmoidf_(g);
        float v=hr[(long)s*96+et*32+o]*g*mk;
        atomicAdd(&agg2[(long)d*32+o], v);
    }
}

// x2 = elu(agg2/deg + xself) ; states = [x1 | x2]
__global__ void k_finalize(const float* __restrict__ x1, const float* __restrict__ agg2,
                           const float* __restrict__ xself, const float* __restrict__ deg,
                           float* __restrict__ states, long n32)
{
    long t = (long)blockIdx.x*blockDim.x + threadIdx.x;
    if(t>=n32) return;
    long n=t>>5; int o=t&31;
    float dg=fmaxf(deg[n],1.f);
    float x2=eluf_(agg2[t]/dg + xself[t]);
    states[n*64+o]=x1[t]; states[n*64+32+o]=x2;
}

// per 16-pair tile: z=[states[u]|states[i]] (16x128 in padded LDS), z@W1 via WMMA,
// relu, dot W2, sigmoid
__global__ void k_head_wmma(const float* __restrict__ states, const int* __restrict__ uidx,
                            const int* __restrict__ iidx, const float* __restrict__ W1,
                            const float* __restrict__ b1, const float* __restrict__ W2,
                            const float* __restrict__ b2, float* __restrict__ out, int ntiles)
{
    __shared__ float Z[WPB*16*ZROW];
    int wave=threadIdx.x>>5, lane=threadIdx.x&31;
    int tile=blockIdx.x*WPB+wave;
    if(tile>=ntiles) return;
    float* Zw = Z + wave*(16*ZROW);
    for(int i=lane;i<2048;i+=32){
        int mm=i>>7, k=i&127;
        int pair=tile*16+mm;
        int node=(k<64)? uidx[pair] : iidx[pair];
        Zw[mm*ZROW+k]=states[(long)node*64 + (k&63)];
    }
    asm volatile("s_wait_dscnt 0" ::: "memory");
    int m=lane&15, half=lane>>4, n=m;
    v16h afr[4];
    #pragma unroll
    for(int kb=0;kb<4;kb++){
        #pragma unroll
        for(int v=0;v<8;v++){
            int k0=kb*32+kpat_(v,half);
            afr[kb][2*v]  =(_Float16)Zw[m*ZROW+k0];
            afr[kb][2*v+1]=(_Float16)Zw[m*ZROW+k0+1];
        }
    }
    // A fragments now in registers; Zw reused for hidden activations z2
    #pragma unroll 1
    for(int nb=0;nb<8;nb++){
        v8f c={};
        #pragma unroll
        for(int kb=0;kb<4;kb++){
            v16h b; int col=nb*16+n;
            #pragma unroll
            for(int v=0;v<8;v++){
                int k0=kb*32+kpat_(v,half);
                b[2*v]  =(_Float16)W1[(long)(k0  )*128+col];
                b[2*v+1]=(_Float16)W1[(long)(k0+1)*128+col];
            }
            c=__builtin_amdgcn_wmma_f32_16x16x32_f16(false,afr[kb],false,b,(short)0,c,false,false);
        }
        float bia=b1[nb*16+n];
        #pragma unroll
        for(int r=0;r<8;r++)
            Zw[(half*8+r)*ZROW + nb*16+n] = fmaxf(c[r]+bia, 0.f);
    }
    asm volatile("s_wait_dscnt 0" ::: "memory");
    if(lane<16){
        float acc=b2[0];
        for(int k=0;k<128;k++) acc += Zw[lane*ZROW+k]*W2[k];
        out[tile*16+lane]=sigmoidf_(acc);
    }
}

extern "C" void kernel_launch(void* const* d_in, const int* in_sizes, int n_in,
                              void* d_out, int out_size, void* d_ws, size_t ws_size,
                              hipStream_t stream)
{
    (void)n_in; (void)out_size; (void)ws_size;
    const float* x      =(const float*)d_in[0];
    const float* efeats =(const float*)d_in[1];
    const float* emask  =(const float*)d_in[2];
    const float* Wn     =(const float*)d_in[3];
    const float* a_src  =(const float*)d_in[4];
    const float* a_dst  =(const float*)d_in[5];
    const float* We     =(const float*)d_in[6];
    const float* be     =(const float*)d_in[7];
    const float* Wae    =(const float*)d_in[8];
    const float* Wrel   =(const float*)d_in[9];
    const float* Wef    =(const float*)d_in[10];
    const float* Wself  =(const float*)d_in[11];
    const float* bself  =(const float*)d_in[12];
    const float* W1     =(const float*)d_in[13];
    const float* b1     =(const float*)d_in[14];
    const float* W2     =(const float*)d_in[15];
    const float* b2     =(const float*)d_in[16];
    const int*   src    =(const int*)d_in[17];
    const int*   dst    =(const int*)d_in[18];
    const int*   etype  =(const int*)d_in[19];
    const int*   uidx   =(const int*)d_in[20];
    const int*   iidx   =(const int*)d_in[21];
    const int N  = in_sizes[0]/4;
    const int E  = in_sizes[1]/64;
    const int Bp = in_sizes[20];

    float* ws=(float*)d_ws;
    size_t off=0;
    float* h      = ws+off; off+=(size_t)N*32;
    float* ssrc   = ws+off; off+=(size_t)N*4;
    float* sdst   = ws+off; off+=(size_t)N*4;
    float* att    = ws+off; off+=(size_t)E*4;
    float* esig   = ws+off; off+=(size_t)E*8;
    float* hr     = ws+off; off+=(size_t)N*96;
    float* xselfp = ws+off; off+=(size_t)N*32;
    float* states = ws+off; off+=(size_t)N*64;
    float* zero0  = ws+off;               // contiguous zero-init region below
    unsigned* mkey=(unsigned*)(ws+off); off+=(size_t)N*4;
    float* ssum   = ws+off; off+=(size_t)N*4;
    float* x1     = ws+off; off+=(size_t)N*32;
    float* agg2   = ws+off; off+=(size_t)N*32;
    float* deg    = ws+off; off+=(size_t)N;
    long zn=(long)N*73;

    dim3 blk(256);
    k_zero<<<(int)((zn+255)/256), blk, 0, stream>>>(zero0, zn);
    k_node_proj<<<(N+255)/256, blk, 0, stream>>>(x,Wn,a_src,a_dst,h,ssrc,sdst,N);
    int etiles=E/16;
    k_edge_wmma<<<(etiles+WPB-1)/WPB, dim3(WPB*32), 0, stream>>>(
        efeats,We,be,Wae,src,dst,ssrc,sdst,esig,att,mkey,etiles,E);
    long e4=(long)E*4;
    k_expsum  <<<(int)((e4+255)/256), blk, 0, stream>>>(att,dst,mkey,ssum,e4);
    k_scatter1<<<(int)((e4+255)/256), blk, 0, stream>>>(att,src,dst,mkey,ssum,emask,h,x1,e4);
    long n32=(long)N*32;
    k_elu_inplace<<<(int)((n32+255)/256), blk, 0, stream>>>(x1,n32);
    int ntiles=(N+15)/16;
    k_node_wmma<<<(ntiles+WPB-1)/WPB, dim3(WPB*32), 0, stream>>>(
        x1,Wrel,Wself,bself,hr,xselfp,ntiles);
    long e8=(long)E*8;
    k_scatter2<<<(int)((e8+255)/256), blk, 0, stream>>>(
        esig,Wef,hr,src,dst,etype,emask,agg2,deg,e8);
    k_finalize<<<(int)((n32+255)/256), blk, 0, stream>>>(x1,agg2,xselfp,deg,states,n32);
    int btiles=Bp/16;
    k_head_wmma<<<(btiles+WPB-1)/WPB, dim3(WPB*32), 0, stream>>>(
        states,uidx,iidx,W1,b1,W2,b2,(float*)d_out,btiles);
}
